// Network_57707180589124
// MI455X (gfx1250) — compile-verified, matrix operand order checked by reading, not run on hardware
//
#include <hip/hip_runtime.h>
#include <hip/hip_bf16.h>
#include <math.h>

typedef __attribute__((ext_vector_type(16))) _Float16 v16h;
typedef __attribute__((ext_vector_type(8)))  float    v8f;

#define E_EDGES   228000
#define N_NODES   12000
#define N_MOL     600
#define EPM       380     // edges per molecule
#define APM       20      // atoms per molecule

// ---------------- workspace layout (bytes, 256-aligned) ----------------
constexpr size_t OFF_EMB = 0;                          // E*64 f16  = 29,184,000
constexpr size_t OFF_SH1 = OFF_EMB + 29184000;         // E*4  f32  =  3,648,000
constexpr size_t OFF_SH2 = OFF_SH1 + 3648000;          // E*8  f32  =  7,296,000
constexpr size_t OFF_W1T = OFF_SH2 + 7296000;          // 3*128*64  f16
constexpr size_t OFF_W2T = OFF_W1T + 49152;            // 3*128*128 f16
constexpr size_t OFF_W3T = OFF_W2T + 98304;            // 3*320*128 f16
constexpr size_t OFF_H1  = OFF_W3T + 245760;           // E*128 f16 = 58,368,000
constexpr size_t OFF_H2  = OFF_H1  + 58368000;         // E*128 f16
constexpr size_t OFF_W   = OFF_H2  + 58368000;         // E*320 f16 = 145,920,000
constexpr size_t OFF_F0  = OFF_W   + 145920000;        // N*64 f32
constexpr size_t OFF_F1  = OFF_F0  + 3072000;          // N*96 f32  (n,c<32,i<3)
constexpr size_t OFF_F2  = OFF_F1  + 4608000;          // N*80 f32  (n,c<16,i<5)

__device__ __forceinline__ float silu_f(float x) { return x / (1.0f + __expf(-x)); }

// ------------- async global->LDS staging (CDNA5 GLOBAL_LOAD_ASYNC_TO_LDS_B128) -------------
#if defined(__HIP_DEVICE_COMPILE__) && defined(__gfx1250__) && \
    __has_builtin(__builtin_amdgcn_global_load_async_to_lds_b128)
#define HAVE_ASYNC_LDS 1
#else
#define HAVE_ASYNC_LDS 0
#endif

#if HAVE_ASYNC_LDS
typedef int v4i_vs __attribute__((vector_size(16)));
typedef __attribute__((address_space(1))) v4i_vs* gv4i_p;
typedef __attribute__((address_space(3))) v4i_vs* lv4i_p;
typedef __attribute__((address_space(1))) char*   gchar_p;
typedef __attribute__((address_space(3))) char*   lchar_p;
#endif

__device__ __forceinline__ void copy_to_lds_16B(void* lds, const void* g, int bytes) {
    const int chunks = bytes >> 4;
#if HAVE_ASYNC_LDS
    gchar_p gp = (gchar_p)g;
    lchar_p lp = (lchar_p)lds;
    for (int i = threadIdx.x; i < chunks; i += blockDim.x) {
        __builtin_amdgcn_global_load_async_to_lds_b128(
            (gv4i_p)(gp + (size_t)i * 16), (lv4i_p)(lp + (size_t)i * 16), 0, 0);
    }
#else
    const uint4* src = (const uint4*)g;
    uint4* dst = (uint4*)lds;
    for (int i = threadIdx.x; i < chunks; i += blockDim.x) dst[i] = src[i];
#endif
}

__device__ __forceinline__ void lds_copy_fence() {
#if HAVE_ASYNC_LDS
#if __has_builtin(__builtin_amdgcn_s_wait_asynccnt)
    __builtin_amdgcn_s_wait_asynccnt(0);
#else
    asm volatile("s_wait_asynccnt 0" ::: "memory");
#endif
#endif
    __syncthreads();
}

// ---------------- kernel: per-edge geometry + radial basis ----------------
__global__ void edge_geom(const float* __restrict__ pos,
                          const int* __restrict__ esrc, const int* __restrict__ edst,
                          _Float16* __restrict__ emb, float* __restrict__ sh1o,
                          float* __restrict__ sh2o) {
    int e = blockIdx.x * blockDim.x + threadIdx.x;
    if (e >= E_EDGES) return;
    int s = esrc[e], d = edst[e];
    float ex = pos[3*s+0] - pos[3*d+0];
    float ey = pos[3*s+1] - pos[3*d+1];
    float ez = pos[3*s+2] - pos[3*d+2];
    float r  = sqrtf(ex*ex + ey*ey + ez*ez);
    float iv = 1.0f / r;
    float x = ex*iv, y = ey*iv, z = ez*iv;
    const float s3  = 1.7320508075688772f;   // sqrt(3)
    const float s15 = 3.8729833462074170f;   // sqrt(15)
    const float s5h = 1.1180339887498949f;   // sqrt(5)/2
    const float s15h= 1.9364916731037085f;   // sqrt(15)/2
    sh1o[4*e+0] = s3*x; sh1o[4*e+1] = s3*y; sh1o[4*e+2] = s3*z; sh1o[4*e+3] = 0.f;
    sh2o[8*e+0] = s15*x*y;
    sh2o[8*e+1] = s15*y*z;
    sh2o[8*e+2] = s5h*(3.0f*z*z - 1.0f);
    sh2o[8*e+3] = s15*x*z;
    sh2o[8*e+4] = s15h*(x*x - y*y);
    sh2o[8*e+5] = 0.f; sh2o[8*e+6] = 0.f; sh2o[8*e+7] = 0.f;
    // soft one-hot radial basis, padded 50 -> 64
    const float step = 10.0f / 51.0f;
    const float cst  = 1.14136f * 7.389056098930650f * 7.0710678118654755f; // 1.14136*e^2*sqrt(50)
    _Float16* eo = emb + (size_t)e * 64;
    #pragma unroll 10
    for (int i = 0; i < 50; ++i) {
        float v = (float)(i + 1) * step;
        float diff = (r - v) / step;
        float a = diff + 1.0f, b = 1.0f - diff;
        float ya = (a > 0.f) ? __expf(-1.0f / a) : 0.0f;
        float yb = (b > 0.f) ? __expf(-1.0f / b) : 0.0f;
        eo[i] = (_Float16)(cst * ya * yb);
    }
    #pragma unroll
    for (int i = 50; i < 64; ++i) eo[i] = (_Float16)0.0f;
}

// ---------------- kernel: convert/transposed f32->f16 MLP weights ----------------
__global__ void convert_weights(const float* __restrict__ mlp1,  // [50][128]
                                const float* __restrict__ mlp2,  // [128][128]
                                const float* __restrict__ mlp3,  // [128][320]
                                _Float16* __restrict__ w1t,      // [128][64]
                                _Float16* __restrict__ w2t,      // [128][128]
                                _Float16* __restrict__ w3t) {    // [320][128]
    int i = blockIdx.x * blockDim.x + threadIdx.x;
    if (i < 128*64)  { int n = i >> 6,  k = i & 63;
        w1t[i] = (_Float16)((k < 50) ? mlp1[k*128 + n] : 0.0f); }
    if (i < 128*128) { int n = i >> 7,  k = i & 127;
        w2t[i] = (_Float16)(mlp2[k*128 + n]); }
    if (i < 320*128) { int n = i >> 7,  k = i & 127;
        w3t[i] = (_Float16)(mlp3[k*320 + n]); }
}

// ---------------- kernel: node feature init ----------------
__global__ void init_nodes(const int* __restrict__ z, const float* __restrict__ Wemb,
                           float* __restrict__ f0, float* __restrict__ f1,
                           float* __restrict__ f2) {
    int n = blockIdx.x * blockDim.x + threadIdx.x;
    if (n >= N_NODES) return;
    int zz = z[n];
    int idx = (zz == 1) ? 0 : (zz == 6) ? 1 : (zz == 7) ? 2 : (zz == 8) ? 3 : 4;
    const float s5 = 2.23606797749979f;
    for (int m = 0; m < 64; ++m) f0[(size_t)n*64 + m] = s5 * Wemb[idx*64 + m];
    for (int i = 0; i < 96; ++i) f1[(size_t)n*96 + i] = 0.0f;
    for (int i = 0; i < 80; ++i) f2[(size_t)n*80 + i] = 0.0f;
}

// ---------------- kernel: tall-skinny WMMA GEMM, Out = act(A[rows,K] @ B[K,N]) ----------------
// A f16 row-major, B supplied transposed f16 [N][K], Out f16 row-major.
// Block = 256 threads (8 waves), each wave owns 16 rows; B staged in LDS (async on CDNA5).
// Requires rows % 16 == 0 (true: E = 228000 = 14250*16), so no per-element store guards.
template <int K, int N, bool ACT>
__global__ void gemm_f16_wmma(const _Float16* __restrict__ A,
                              const _Float16* __restrict__ Bt,
                              _Float16* __restrict__ Out, int rows) {
    extern __shared__ _Float16 sB[];              // N*K halves
    copy_to_lds_16B(sB, Bt, N * K * 2);
    lds_copy_fence();

    const int wave = threadIdx.x >> 5;
    const int lane = threadIdx.x & 31;
    const int r0 = blockIdx.x * 128 + wave * 16;
    if (r0 >= rows) return;                        // wave-uniform exit; EXEC stays full
    const int hi = lane >> 4;                      // 0: K 0..15, 1: K 16..31 of the k-tile
    const int m  = lane & 15;
    const int r  = r0 + m;

    constexpr int NK = K / 32;
    v16h afrag[NK];
    #pragma unroll
    for (int kt = 0; kt < NK; ++kt)
        afrag[kt] = *(const v16h*)(A + (size_t)r * K + kt * 32 + hi * 16);

    _Float16* outp = Out + (size_t)(r0 + 8 * hi) * N + m;   // C rows for this lane
    constexpr int NN = N / 16;
    #pragma unroll 4
    for (int nt = 0; nt < NN; ++nt) {
        v8f acc = {};
        const int n = nt * 16 + m;                 // B-fragment column for this lane
        #pragma unroll
        for (int kt = 0; kt < NK; ++kt) {
            v16h bfrag = *(const v16h*)(sB + (size_t)n * K + kt * 32 + hi * 16);
            acc = __builtin_amdgcn_wmma_f32_16x16x32_f16(
                false, afrag[kt], false, bfrag, (short)0, acc, false, false);
        }
        #pragma unroll
        for (int v = 0; v < 8; ++v) {              // C layout: M = v + 8*hi, Ncol = m
            float xv = acc[v];
            if (ACT) xv = silu_f(xv);
            outp[(size_t)v * N + nt * 16] = (_Float16)xv;
        }
    }
}

// ---------------- kernel: per-molecule message + LDS scatter + node update ----------------
__global__ void layer_message_update(
    const _Float16* __restrict__ w,   // [E][320]
    const float* __restrict__ sh1,    // [E][4]
    const float* __restrict__ sh2,    // [E][8]
    const int* __restrict__ esrc, const int* __restrict__ edst,
    float* __restrict__ f0, float* __restrict__ f1, float* __restrict__ f2,
    const float* __restrict__ Wm0, const float* __restrict__ Ws0,
    const float* __restrict__ Wg1, const float* __restrict__ Wg2,
    const float* __restrict__ Wm1, const float* __restrict__ Ws1,
    const float* __restrict__ Wm2, const float* __restrict__ Ws2) {
    extern __shared__ float smem[];
    float* lf0 = smem;              // 20*64  = 1280
    float* lf1 = lf0 + 1280;        // 20*96  = 1920
    float* lf2 = lf1 + 1920;        // 20*80  = 1600
    float* la0 = lf2 + 1600;        // 20*112 = 2240
    float* la1 = la0 + 2240;        // 20*384 = 7680
    float* la2 = la1 + 7680;        // 20*400 = 8000
    float* ls  = la2 + 8000;        // 20*64  = 1280   -> 24000 floats = 96000 B

    const int mol = blockIdx.x;
    const int nbase = mol * APM;
    copy_to_lds_16B(lf0, f0 + (size_t)nbase * 64, 1280 * 4);
    copy_to_lds_16B(lf1, f1 + (size_t)nbase * 96, 1920 * 4);
    copy_to_lds_16B(lf2, f2 + (size_t)nbase * 80, 1600 * 4);
    for (int i = threadIdx.x; i < 2240 + 7680 + 8000; i += blockDim.x) la0[i] = 0.0f;
    lds_copy_fence();

    const float inv_nn = 0.22360679774997896f;   // 1/sqrt(20)
    const int ebase = mol * EPM;
    // 5 balanced channel-groups per edge (~160-192 LDS atomics each)
    for (int t = threadIdx.x; t < EPM * 5; t += blockDim.x) {
        const int el = t / 5;
        const int part = t - el * 5;
        const int e = ebase + el;
        const int sl = esrc[e] - nbase;
        const int dl = edst[e] - nbase;
        const _Float16* we = w + (size_t)e * 320;
        const float* F0 = lf0 + sl * 64;
        const float* F1 = lf1 + sl * 96;
        const float* F2 = lf2 + sl * 80;
        float s1[3], s2[5];
        s1[0] = sh1[4*e+0]; s1[1] = sh1[4*e+1]; s1[2] = sh1[4*e+2];
        #pragma unroll
        for (int i = 0; i < 5; ++i) s2[i] = sh2[8*e+i];
        if (part == 0) {               // msg0 (112) + msg2/w7 (80)
            float* A0 = la0 + dl * 112;
            for (int c = 0; c < 64; ++c)
                atomicAdd(&A0[c], inv_nn * (float)we[c] * F0[c]);
            for (int c = 0; c < 32; ++c) {
                float dd = F1[c*3+0]*s1[0] + F1[c*3+1]*s1[1] + F1[c*3+2]*s1[2];
                atomicAdd(&A0[64+c], inv_nn * (float)we[64+c] * dd);
            }
            for (int c = 0; c < 16; ++c) {
                float dd = 0.f;
                #pragma unroll
                for (int i = 0; i < 5; ++i) dd += F2[c*5+i] * s2[i];
                atomicAdd(&A0[96+c], inv_nn * (float)we[96+c] * dd);
            }
            float* A2 = la2 + dl * 400;
            for (int c = 0; c < 16; ++c) {
                float tt = inv_nn * (float)we[304+c];
                #pragma unroll
                for (int i = 0; i < 5; ++i) atomicAdd(&A2[(64+c)*5+i], tt * F2[c*5+i]);
            }
        } else if (part == 1) {        // msg1: w3*f0 (x) sh1, 64 ch
            float* A1 = la1 + dl * 384;
            for (int c = 0; c < 64; ++c) {
                float tt = inv_nn * (float)we[112+c] * F0[c];
                #pragma unroll
                for (int i = 0; i < 3; ++i) atomicAdd(&A1[c*3+i], tt * s1[i]);
            }
        } else if (part == 2) {        // msg1: w4*f1 | w5*cross(f1,sh1), 64 ch
            float* A1 = la1 + dl * 384;
            for (int c = 0; c < 32; ++c) {
                float tt = inv_nn * (float)we[176+c];
                #pragma unroll
                for (int i = 0; i < 3; ++i) atomicAdd(&A1[(64+c)*3+i], tt * F1[c*3+i]);
            }
            for (int c = 0; c < 32; ++c) {
                float tt = inv_nn * (float)we[208+c];
                float cx = F1[c*3+1]*s1[2] - F1[c*3+2]*s1[1];
                float cy = F1[c*3+2]*s1[0] - F1[c*3+0]*s1[2];
                float cz = F1[c*3+0]*s1[1] - F1[c*3+1]*s1[0];
                atomicAdd(&A1[(96+c)*3+0], tt * cx);
                atomicAdd(&A1[(96+c)*3+1], tt * cy);
                atomicAdd(&A1[(96+c)*3+2], tt * cz);
            }
        } else {                       // parts 3/4: msg2 w6*f0 (x) sh2, 32 ch each
            const int c0 = (part == 3) ? 0 : 32;
            float* A2 = la2 + dl * 400;
            for (int c = c0; c < c0 + 32; ++c) {
                float tt = inv_nn * (float)we[240+c] * F0[c];
                #pragma unroll
                for (int i = 0; i < 5; ++i) atomicAdd(&A2[c*5+i], tt * s2[i]);
            }
        }
    }
    __syncthreads();

    // s = a0 @ Wm0 + f0 @ Ws0 ; f0_new = silu(s)
    for (int o = threadIdx.x; o < APM * 64; o += blockDim.x) {
        int n = o >> 6, d = o & 63;
        float acc = 0.0f;
        const float* A0 = la0 + n * 112;
        for (int c = 0; c < 112; ++c) acc += A0[c] * Wm0[c*64 + d];
        const float* F0 = lf0 + n * 64;
        for (int c = 0; c < 64; ++c)  acc += F0[c] * Ws0[c*64 + d];
        ls[o] = acc;
        f0[(size_t)(nbase + n)*64 + d] = silu_f(acc);
    }
    __syncthreads();
    // f1_new = sigmoid(s@Wg1) * (a1@Wm1 + f1@Ws1)  -- one thread per (n,d), 3 comps
    for (int o = threadIdx.x; o < APM * 32; o += blockDim.x) {
        int n = o >> 5, d = o & 31;
        const float* S = ls + n * 64;
        float g = 0.0f;
        for (int c = 0; c < 64; ++c) g += S[c] * Wg1[c*32 + d];
        g = 1.0f / (1.0f + __expf(-g));
        float acc0 = 0.f, acc1 = 0.f, acc2 = 0.f;
        const float* A1 = la1 + n * 384;
        for (int c = 0; c < 128; ++c) {
            float wv = Wm1[c*32 + d];
            acc0 += A1[c*3+0] * wv; acc1 += A1[c*3+1] * wv; acc2 += A1[c*3+2] * wv;
        }
        const float* F1p = lf1 + n * 96;
        for (int c = 0; c < 32; ++c) {
            float wv = Ws1[c*32 + d];
            acc0 += F1p[c*3+0] * wv; acc1 += F1p[c*3+1] * wv; acc2 += F1p[c*3+2] * wv;
        }
        float* o1 = f1 + (size_t)(nbase + n)*96 + d*3;
        o1[0] = g * acc0; o1[1] = g * acc1; o1[2] = g * acc2;
    }
    // f2_new = sigmoid(s@Wg2) * (a2@Wm2 + f2@Ws2)  -- one thread per (n,d), 5 comps
    for (int o = threadIdx.x; o < APM * 16; o += blockDim.x) {
        int n = o >> 4, d = o & 15;
        const float* S = ls + n * 64;
        float g = 0.0f;
        for (int c = 0; c < 64; ++c) g += S[c] * Wg2[c*16 + d];
        g = 1.0f / (1.0f + __expf(-g));
        float acc[5] = {0.f, 0.f, 0.f, 0.f, 0.f};
        const float* A2p = la2 + n * 400;
        for (int c = 0; c < 80; ++c) {
            float wv = Wm2[c*16 + d];
            #pragma unroll
            for (int i = 0; i < 5; ++i) acc[i] += A2p[c*5+i] * wv;
        }
        const float* F2p = lf2 + n * 80;
        for (int c = 0; c < 16; ++c) {
            float wv = Ws2[c*16 + d];
            #pragma unroll
            for (int i = 0; i < 5; ++i) acc[i] += F2p[c*5+i] * wv;
        }
        float* o2 = f2 + (size_t)(nbase + n)*80 + d*5;
        #pragma unroll
        for (int i = 0; i < 5; ++i) o2[i] = g * acc[i];
    }
}

// ---------------- kernel: readout, one wave32 per molecule ----------------
__global__ void readout(const float* __restrict__ f0, const float* __restrict__ Wout,
                        float* __restrict__ y) {
    const int mol = blockIdx.x;
    const int lane = threadIdx.x;
    float val = 0.0f;
    if (lane < APM) {
        const float* F = f0 + (size_t)(mol * APM + lane) * 64;
        float o0 = 0.f, o1 = 0.f;
        for (int c = 0; c < 64; ++c) { o0 += F[c] * Wout[c*2]; o1 += F[c] * Wout[c*2+1]; }
        val = (o0 + 0.5f * o1 * o1) * 0.22360679774997896f;  // 1/sqrt(20)
    }
    #pragma unroll
    for (int off = 16; off > 0; off >>= 1) val += __shfl_down(val, off, 32);
    if (lane == 0) y[mol] = val;
}

// ---------------- host ----------------
extern "C" void kernel_launch(void* const* d_in, const int* in_sizes, int n_in,
                              void* d_out, int out_size, void* d_ws, size_t ws_size,
                              hipStream_t stream) {
    (void)in_sizes; (void)n_in; (void)out_size; (void)ws_size;
    // params pytree flattens sorted-key: W_embed, W_out, layers[l]{Wg1,Wg2,Wm0,Wm1,Wm2,Ws0,Ws1,Ws2,mlp1,mlp2,mlp3}
    const float* pos  = (const float*)d_in[0];
    const float* Wemb = (const float*)d_in[1];
    const float* Wout = (const float*)d_in[2];
    const float *Wg1[3], *Wg2[3], *Wm0[3], *Wm1[3], *Wm2[3], *Ws0[3], *Ws1[3], *Ws2[3];
    const float *mlp1[3], *mlp2[3], *mlp3[3];
    for (int l = 0; l < 3; ++l) {
        int b = 3 + l * 11;
        Wg1[l]  = (const float*)d_in[b+0];
        Wg2[l]  = (const float*)d_in[b+1];
        Wm0[l]  = (const float*)d_in[b+2];
        Wm1[l]  = (const float*)d_in[b+3];
        Wm2[l]  = (const float*)d_in[b+4];
        Ws0[l]  = (const float*)d_in[b+5];
        Ws1[l]  = (const float*)d_in[b+6];
        Ws2[l]  = (const float*)d_in[b+7];
        mlp1[l] = (const float*)d_in[b+8];
        mlp2[l] = (const float*)d_in[b+9];
        mlp3[l] = (const float*)d_in[b+10];
    }
    const int* z    = (const int*)d_in[36];
    const int* esrc = (const int*)d_in[38];
    const int* edst = (const int*)d_in[39];
    float* y = (float*)d_out;

    char* ws = (char*)d_ws;
    _Float16* emb = (_Float16*)(ws + OFF_EMB);
    float*    sh1 = (float*)(ws + OFF_SH1);
    float*    sh2 = (float*)(ws + OFF_SH2);
    _Float16* h1  = (_Float16*)(ws + OFF_H1);
    _Float16* h2  = (_Float16*)(ws + OFF_H2);
    _Float16* wbuf= (_Float16*)(ws + OFF_W);
    float*    f0  = (float*)(ws + OFF_F0);
    float*    f1  = (float*)(ws + OFF_F1);
    float*    f2  = (float*)(ws + OFF_F2);

    // allow large dynamic LDS (96 KB msg kernel, 80 KB gemm3); ignore errors
    (void)hipFuncSetAttribute((const void*)gemm_f16_wmma<128,320,false>,
                              hipFuncAttributeMaxDynamicSharedMemorySize, 81920);
    (void)hipFuncSetAttribute((const void*)layer_message_update,
                              hipFuncAttributeMaxDynamicSharedMemorySize, 96000);

    edge_geom<<<(E_EDGES + 255) / 256, 256, 0, stream>>>(pos, esrc, edst, emb, sh1, sh2);
    init_nodes<<<(N_NODES + 255) / 256, 256, 0, stream>>>(z, Wemb, f0, f1, f2);
    for (int l = 0; l < 3; ++l) {
        _Float16* w1t = (_Float16*)(ws + OFF_W1T) + (size_t)l * 128 * 64;
        _Float16* w2t = (_Float16*)(ws + OFF_W2T) + (size_t)l * 128 * 128;
        _Float16* w3t = (_Float16*)(ws + OFF_W3T) + (size_t)l * 320 * 128;
        convert_weights<<<(320*128 + 255) / 256, 256, 0, stream>>>(
            mlp1[l], mlp2[l], mlp3[l], w1t, w2t, w3t);
    }

    const int gtiles = (E_EDGES + 127) / 128;
    for (int l = 0; l < 3; ++l) {
        _Float16* w1t = (_Float16*)(ws + OFF_W1T) + (size_t)l * 128 * 64;
        _Float16* w2t = (_Float16*)(ws + OFF_W2T) + (size_t)l * 128 * 128;
        _Float16* w3t = (_Float16*)(ws + OFF_W3T) + (size_t)l * 320 * 128;
        gemm_f16_wmma<64, 128, true ><<<gtiles, 256, 128*64*2,  stream>>>(emb, w1t, h1, E_EDGES);
        gemm_f16_wmma<128,128, true ><<<gtiles, 256, 128*128*2, stream>>>(h1,  w2t, h2, E_EDGES);
        gemm_f16_wmma<128,320, false><<<gtiles, 256, 320*128*2, stream>>>(h2,  w3t, wbuf, E_EDGES);
        layer_message_update<<<N_MOL, 256, 96000, stream>>>(
            wbuf, sh1, sh2, esrc, edst, f0, f1, f2,
            Wm0[l], Ws0[l], Wg1[l], Wg2[l], Wm1[l], Ws1[l], Wm2[l], Ws2[l]);
    }
    readout<<<N_MOL, 32, 0, stream>>>(f0, Wout, y);
}
